// Attention_23888608101232
// MI455X (gfx1250) — matrix operand order repaired and checked by reading, not compiled, not run on hardware
//
#include <hip/hip_runtime.h>

typedef __attribute__((ext_vector_type(16))) _Float16 v16h;
typedef __attribute__((ext_vector_type(8)))  _Float16 v8h;
typedef __attribute__((ext_vector_type(4)))  _Float16 v4h;
typedef __attribute__((ext_vector_type(8)))  float    v8f;

#define WMMA_F16(A, B, C) \
  __builtin_amdgcn_wmma_f32_16x16x32_f16(false, (A), false, (B), (short)0, (C), false, false)

// ---- fragment loaders (16-bit A 16x32 and B 32x16 layouts, contiguous k order: k = 16*(lane/16) + s) ----

// A 16x32: lanes<16 -> row=lane, K=0..15; lanes>=16 -> same rows, K=16..31.
__device__ __forceinline__ v16h load_a16(const _Float16* __restrict__ base, int lda,
                                         int row, int kb, int lane) {
  const int kbase = (lane < 16) ? 0 : 16;
  const _Float16* p = base + (size_t)row * lda + kb + kbase;
  v8h lo = *(const v8h*)p;          // K = kbase..kbase+7
  v8h hi = *(const v8h*)(p + 8);    // K = kbase+8..kbase+15
  return __builtin_shufflevector(lo, hi, 0,1,2,3,4,5,6,7,8,9,10,11,12,13,14,15);
}

// B 32x16 where B[k][n] = W[n][k] (W row-major over K): lane col = lane&15,
// lanes<16 cover K 0..15, lanes>=16 cover K 16..31 -> one contiguous 32B load.
__device__ __forceinline__ v16h load_bT16(const _Float16* __restrict__ base, int ldb,
                                          int col, int kb, int lane) {
  const int koff = (lane < 16) ? 0 : 16;
  return *(const v16h*)(base + (size_t)col * ldb + kb + koff);
}

// ---- elementwise f32 -> f16 convert ----
__global__ void __launch_bounds__(256) cvt_f32_f16(const float* __restrict__ src,
                                                   _Float16* __restrict__ dst, int n) {
  const int i = (blockIdx.x * 256 + threadIdx.x) * 4;
  if (i + 3 < n) {
    float4 v = *(const float4*)(src + i);
    v4h o = { (_Float16)v.x, (_Float16)v.y, (_Float16)v.z, (_Float16)v.w };
    *(v4h*)(dst + i) = o;
  }
}

// ---- QKV projection: [8192,1024] @ [3072,1024]^T + bias -> scatter q/k/vT per head ----
// Wave tile: 32(M) x 64(N); 2 A-fragments reused over 4 B-fragments -> 8 WMMA per k-step.
__global__ void __launch_bounds__(256) qkv_gemm_kernel(
    const _Float16* __restrict__ xh, const _Float16* __restrict__ wh,
    const float* __restrict__ bias,
    _Float16* __restrict__ qh, _Float16* __restrict__ kh, _Float16* __restrict__ vth) {
  const int lane = threadIdx.x & 31;
  const int wv   = threadIdx.x >> 5;
  const int wid  = blockIdx.x * 8 + wv;          // 12288 waves
  const int NT   = 3072 / 64;                    // 48 n-tiles of 64
  const int mi = wid / NT, ni = wid % NT;
  const int m0 = mi * 32, n0 = ni * 64;
  const int ncol = lane & 15, hf = lane >> 4;

  v8f acc[2][4] = {};
  for (int kb = 0; kb < 1024; kb += 32) {
    v16h a0 = load_a16(xh, 1024, m0 + ncol,      kb, lane);
    v16h a1 = load_a16(xh, 1024, m0 + 16 + ncol, kb, lane);
#pragma unroll
    for (int t = 0; t < 4; ++t) {
      v16h b = load_bT16(wh, 1024, n0 + t * 16 + ncol, kb, lane);
      acc[0][t] = WMMA_F16(a0, b, acc[0][t]);
      acc[1][t] = WMMA_F16(a1, b, acc[1][t]);
    }
  }

  // n0 is 64-aligned -> whole wave tile lies in one (q/k/v, head) slot.
  const int which = n0 >> 10;             // 0=q, 1=k, 2=v
  const int h     = (n0 & 1023) >> 6;     // head
  const int bb    = m0 >> 11;             // batch (m0 32-aligned, 2048 % 32 == 0)
  const int tok0  = m0 & 2047;
  const size_t bh = (size_t)(bb * 16 + h);
#pragma unroll
  for (int g = 0; g < 2; ++g) {
#pragma unroll
    for (int t = 0; t < 4; ++t) {
      const int j = n0 + t * 16 + ncol;
      const float bsv = bias[j];
      const int e = j & 63;
#pragma unroll
      for (int r = 0; r < 8; ++r) {
        const int tok = tok0 + g * 16 + r + hf * 8;  // C-layout rows
        const _Float16 hv = (_Float16)(acc[g][t][r] + bsv);
        if (which == 0)      qh[(bh * 2048 + tok) * 64 + e] = hv;
        else if (which == 1) kh[(bh * 2048 + tok) * 64 + e] = hv;
        else                 vth[(bh * 64 + e) * 2048 + tok] = hv;  // V transposed
      }
    }
  }
}

// ---- flash attention: one wave per 32-query tile, online softmax over 32-key chunks ----
// K and V fragments are loaded once per chunk and shared by both 16-query sub-tiles.
__global__ void __launch_bounds__(256) attn_kernel(
    const _Float16* __restrict__ qh, const _Float16* __restrict__ kh,
    const _Float16* __restrict__ vth, _Float16* __restrict__ attn) {
  __shared__ _Float16 pb[8][32][32];                  // per-wave P staging (C->A layout)
  const int lane = threadIdx.x & 31;
  const int wv   = threadIdx.x >> 5;
  const int wid  = blockIdx.x * 8 + wv;               // 4096 waves
  const int qt   = wid & 63;                          // 2048/32 query tiles per (b,h)
  const int bh   = wid >> 6;                          // 0..63 (b*16+h)
  const int b    = bh >> 4, h = bh & 15;
  const int q0   = qt * 32;
  const int ncol = lane & 15, hf = lane >> 4;
  const float SCALE = 0.125f;                         // 64^-0.5

  const _Float16* Qp = qh  + (size_t)bh * 2048 * 64;
  const _Float16* Kp = kh  + (size_t)bh * 2048 * 64;
  const _Float16* Vp = vth + (size_t)bh * 64 * 2048;  // [64][2048]

  v16h aq[2][2];
#pragma unroll
  for (int g = 0; g < 2; ++g) {
    aq[g][0] = load_a16(Qp, 64, q0 + g * 16 + ncol, 0, lane);
    aq[g][1] = load_a16(Qp, 64, q0 + g * 16 + ncol, 32, lane);
  }

  float mi[2][8], li[2][8];
#pragma unroll
  for (int g = 0; g < 2; ++g)
#pragma unroll
    for (int r = 0; r < 8; ++r) { mi[g][r] = -1e30f; li[g][r] = 0.f; }
  v8f o[2][4] = {};

  for (int kb = 0; kb < 2048; kb += 32) {
    // S = Q*K^T for 32 keys x 32 queries: 4 S tiles, K frags shared across query tiles
    v16h bk[2][2];
#pragma unroll
    for (int kt = 0; kt < 2; ++kt) {
      bk[kt][0] = *(const v16h*)(Kp + (size_t)(kb + kt * 16 + ncol) * 64 + hf * 16);
      bk[kt][1] = *(const v16h*)(Kp + (size_t)(kb + kt * 16 + ncol) * 64 + 32 + hf * 16);
    }
    v8f s[2][2] = {};
#pragma unroll
    for (int g = 0; g < 2; ++g)
#pragma unroll
      for (int kt = 0; kt < 2; ++kt) {
        s[g][kt] = WMMA_F16(aq[g][0], bk[kt][0], s[g][kt]);
        s[g][kt] = WMMA_F16(aq[g][1], bk[kt][1], s[g][kt]);
      }

    // online softmax; row r of sub-tile g lives in VGPR r across each 16-lane half
#pragma unroll
    for (int g = 0; g < 2; ++g) {
#pragma unroll
      for (int r = 0; r < 8; ++r) {
        const float v0 = s[g][0][r] * SCALE;
        const float v1 = s[g][1][r] * SCALE;
        float mr = fmaxf(v0, v1);
        mr = fmaxf(mr, __shfl_xor(mr, 1, 32));
        mr = fmaxf(mr, __shfl_xor(mr, 2, 32));
        mr = fmaxf(mr, __shfl_xor(mr, 4, 32));
        mr = fmaxf(mr, __shfl_xor(mr, 8, 32));
        const float mnew  = fmaxf(mi[g][r], mr);
        const float alpha = __expf(mi[g][r] - mnew);
        const float p0 = __expf(v0 - mnew);
        const float p1 = __expf(v1 - mnew);
        float rs = p0 + p1;
        rs += __shfl_xor(rs, 1, 32);
        rs += __shfl_xor(rs, 2, 32);
        rs += __shfl_xor(rs, 4, 32);
        rs += __shfl_xor(rs, 8, 32);
        li[g][r] = li[g][r] * alpha + rs;
        mi[g][r] = mnew;
#pragma unroll
        for (int t = 0; t < 4; ++t) o[g][t][r] *= alpha;
        const int row = g * 16 + r + hf * 8;
        pb[wv][row][ncol]      = (_Float16)p0;
        pb[wv][row][16 + ncol] = (_Float16)p1;
      }
    }
    // same-wave LDS ops are in-order; wait for stores, then reload in A layout
    asm volatile("s_wait_dscnt 0" ::: "memory");
    v16h ap[2];
#pragma unroll
    for (int g = 0; g < 2; ++g) {
      const _Float16* pp = &pb[wv][g * 16 + ncol][hf * 16];
      v8h lo = *(const v8h*)pp;
      v8h hi = *(const v8h*)(pp + 8);
      ap[g] = __builtin_shufflevector(lo, hi, 0,1,2,3,4,5,6,7,8,9,10,11,12,13,14,15);
    }
    // O += P * V  (V fragment loaded once, shared by both query sub-tiles)
#pragma unroll
    for (int t = 0; t < 4; ++t) {
      v16h bv = *(const v16h*)(Vp + (size_t)(t * 16 + ncol) * 2048 + kb + hf * 16);
      o[0][t] = WMMA_F16(ap[0], bv, o[0][t]);
      o[1][t] = WMMA_F16(ap[1], bv, o[1][t]);
    }
  }

  // normalize and store f16 into [b][tok][h*64 + e] for the output GEMM
#pragma unroll
  for (int g = 0; g < 2; ++g)
#pragma unroll
    for (int r = 0; r < 8; ++r) {
      const float inv = 1.0f / li[g][r];
      const int tok = q0 + g * 16 + r + hf * 8;
#pragma unroll
      for (int t = 0; t < 4; ++t) {
        attn[((size_t)(b * 2048 + tok)) * 1024 + h * 64 + t * 16 + ncol] =
            (_Float16)(o[g][t][r] * inv);
      }
    }
}

// ---- output projection: [8192,1024] @ [1024,1024]^T + bias -> f32 d_out ----
// Wave tile: 32(M) x 64(N).
__global__ void __launch_bounds__(256) out_gemm_kernel(
    const _Float16* __restrict__ ah, const _Float16* __restrict__ wh,
    const float* __restrict__ bias, float* __restrict__ out) {
  const int lane = threadIdx.x & 31;
  const int wv   = threadIdx.x >> 5;
  const int wid  = blockIdx.x * 8 + wv;     // 4096 waves
  const int NT   = 1024 / 64;               // 16
  const int mi = wid / NT, ni = wid % NT;
  const int m0 = mi * 32, n0 = ni * 64;
  const int ncol = lane & 15, hf = lane >> 4;

  v8f acc[2][4] = {};
  for (int kb = 0; kb < 1024; kb += 32) {
    v16h a0 = load_a16(ah, 1024, m0 + ncol,      kb, lane);
    v16h a1 = load_a16(ah, 1024, m0 + 16 + ncol, kb, lane);
#pragma unroll
    for (int t = 0; t < 4; ++t) {
      v16h b = load_bT16(wh, 1024, n0 + t * 16 + ncol, kb, lane);
      acc[0][t] = WMMA_F16(a0, b, acc[0][t]);
      acc[1][t] = WMMA_F16(a1, b, acc[1][t]);
    }
  }
#pragma unroll
  for (int g = 0; g < 2; ++g)
#pragma unroll
    for (int t = 0; t < 4; ++t) {
      const int n = n0 + t * 16 + ncol;
      const float bs = bias[n];
#pragma unroll
      for (int r = 0; r < 8; ++r) {
        const int m = m0 + g * 16 + r + hf * 8;
        out[(size_t)m * 1024 + n] = acc[g][t][r] + bs;
      }
    }
}

extern "C" void kernel_launch(void* const* d_in, const int* in_sizes, int n_in,
                              void* d_out, int out_size, void* d_ws, size_t ws_size,
                              hipStream_t stream) {
  const float* x     = (const float*)d_in[0];   // [4,2048,1024]
  const float* qkv_w = (const float*)d_in[1];   // [3072,1024]
  const float* qkv_b = (const float*)d_in[2];   // [3072]
  const float* out_w = (const float*)d_in[3];   // [1024,1024]
  const float* out_b = (const float*)d_in[4];   // [1024]
  float* out = (float*)d_out;

  _Float16* ws    = (_Float16*)d_ws;
  _Float16* xh    = ws;                          // 8,388,608 halves
  _Float16* wqkvh = ws + 8388608;                // 3,145,728
  _Float16* wouth = ws + 11534336;               // 1,048,576
  _Float16* qh    = ws + 12582912;               // 8,388,608  [bh][n][64]
  _Float16* kh    = ws + 20971520;               // 8,388,608  [bh][n][64]
  _Float16* vth   = ws + 29360128;               // 8,388,608  [bh][64][n]
  _Float16* ah    = ws + 37748736;               // 8,388,608  [b*n][1024]

  cvt_f32_f16<<<8192, 256, 0, stream>>>(x, xh, 8388608);
  cvt_f32_f16<<<3072, 256, 0, stream>>>(qkv_w, wqkvh, 3145728);
  cvt_f32_f16<<<1024, 256, 0, stream>>>(out_w, wouth, 1048576);

  // 256 M-tiles(32) x 48 N-tiles(64) = 12288 waves
  qkv_gemm_kernel<<<1536, 256, 0, stream>>>(xh, wqkvh, qkv_b, qh, kh, vth);
  // 64 bh x 64 query-tiles(32) = 4096 waves
  attn_kernel<<<512, 256, 0, stream>>>(qh, kh, vth, ah);
  // 256 M-tiles(32) x 16 N-tiles(64) = 4096 waves
  out_gemm_kernel<<<512, 256, 0, stream>>>(ah, wouth, out_b, out);
}
